// MultiHeadSelfAttention_54812372632307
// MI455X (gfx1250) — matrix operand order repaired and checked by reading, not compiled, not run on hardware
//
#include <hip/hip_runtime.h>
#include <hip/hip_bf16.h>
#include <math.h>

// ---------------------------------------------------------------------------
// Types
// ---------------------------------------------------------------------------
typedef _Float16 f16;
typedef __attribute__((ext_vector_type(16))) _Float16 v16h;
typedef __attribute__((ext_vector_type(8)))  _Float16 v8h;
typedef __attribute__((ext_vector_type(2)))  _Float16 v2h;
typedef __attribute__((ext_vector_type(8)))  float    v8f;

// Problem constants (from reference): B=4, T=2048, C=1024, H=16, hd=64
#define BB   4
#define TT   2048
#define CC   1024
#define HH   16
#define HD   64
#define C3   (3*CC)
#define MROWS (BB*TT)          // 8192

// softmax scale folded with log2(e): exp(0.125*s) == exp2(0.18033689*s)
#define QSCALE_LOG2E 0.18033688f

// ---------------------------------------------------------------------------
// WMMA helpers (CDNA5, wave32).  D = A(16x32 f16) * B(32x16 f16) + C(16x16 f32)
// ---------------------------------------------------------------------------
__device__ __forceinline__ v8f wmma_f16(v16h a, v16h b, v8f c) {
  // (neg_a, A, neg_b, B, c_mod, C, reuse_a, reuse_b)
  return __builtin_amdgcn_wmma_f32_16x16x32_f16(false, a, false, b, (short)0, c,
                                                false, false);
}

// A-fragment, 16x32 f16 (M x K), from row-major storage with row stride ldk.
// Lane L (0-15): row M=L, VGPR0-3 = K[half..half+7], VGPR4-7 = K[16+half..+7]
// where half = 0 for lanes 0-15, 8 for lanes 16-31.  (ISA 7.12.2, 16-bit A 16x32)
__device__ __forceinline__ v16h load_a_frag(const f16* base, int ldk, int lane) {
  int row  = lane & 15;
  int half = (lane >> 4) << 3;               // 0 or 8
  const f16* p = base + (size_t)row * ldk + half;
  v16h a;
  *((v8h*)&a)     = *(const v8h*)p;          // K = half .. half+7
  *((v8h*)&a + 1) = *(const v8h*)(p + 16);   // K = 16+half .. 16+half+7
  return a;
}

// B-fragment, 32x16 f16 (K x N), from COLUMN-major storage (each column's K
// values contiguous) with column stride ldk.
// Lane L (0-15): col N=L, K=0..7,16..23; lanes 16-31: col N=L-16, K=8..15,24..31.
__device__ __forceinline__ v16h load_b_frag(const f16* base, int ldk, int lane) {
  int col   = lane & 15;
  int khalf = (lane >> 4) << 3;              // 0 or 8
  const f16* p = base + (size_t)col * ldk + khalf;
  v16h b;
  *((v8h*)&b)     = *(const v8h*)p;          // K = khalf .. khalf+7
  *((v8h*)&b + 1) = *(const v8h*)(p + 16);   // K = 16+khalf .. 16+khalf+7
  return b;
}

// ---------------------------------------------------------------------------
// DPP16 butterfly reductions across the 16-lane row (VALU pipe, no DS).
// old=0 + bound_ctrl=1: no dependence on the old value -> no copy chains.
// All controls fetch from active lanes, so the 0-fallback never appears.
// ---------------------------------------------------------------------------
template <int CTRL>
__device__ __forceinline__ float dpp_xf(float x) {
  int xi = __builtin_bit_cast(int, x);
  int r  = __builtin_amdgcn_update_dpp(0, xi, CTRL, 0xF, 0xF, true);
  return __builtin_bit_cast(float, r);
}
#define DPP_QUAD_XOR1   0xB1   // quad_perm(1,0,3,2)
#define DPP_QUAD_XOR2   0x4E   // quad_perm(2,3,0,1)
#define DPP_HALF_MIRROR 0x141  // lane ^ 7 within 8
#define DPP_ROW_MIRROR  0x140  // lane ^ 15 within 16

__device__ __forceinline__ float row16_max(float v) {
  v = fmaxf(v, dpp_xf<DPP_QUAD_XOR1>(v));
  v = fmaxf(v, dpp_xf<DPP_QUAD_XOR2>(v));
  v = fmaxf(v, dpp_xf<DPP_HALF_MIRROR>(v));
  v = fmaxf(v, dpp_xf<DPP_ROW_MIRROR>(v));
  return v;
}
__device__ __forceinline__ float row16_sum(float v) {
  v += dpp_xf<DPP_QUAD_XOR1>(v);
  v += dpp_xf<DPP_QUAD_XOR2>(v);
  v += dpp_xf<DPP_HALF_MIRROR>(v);
  v += dpp_xf<DPP_ROW_MIRROR>(v);
  return v;
}

// ---------------------------------------------------------------------------
// CDNA5 async copy: global -> LDS, 16B per lane, tracked by ASYNCcnt.
// ---------------------------------------------------------------------------
__device__ __forceinline__ void async_ld16(unsigned lds_off, const void* saddr,
                                           int voff) {
  asm volatile("global_load_async_to_lds_b128 %0, %1, %2"
               :: "v"(lds_off), "v"(voff), "s"(saddr)
               : "memory");
}
__device__ __forceinline__ unsigned lds_addr(const void* p) {
  return (unsigned)(unsigned long long)p;    // flat LDS: addr[31:0] = LDS offset
}
#define WAIT_ASYNC_3() asm volatile("s_wait_asynccnt 0x3" ::: "memory")
#define WAIT_ASYNC_0() asm volatile("s_wait_asynccnt 0x0" ::: "memory")

// ---------------------------------------------------------------------------
// Elementwise f32 -> f16
// ---------------------------------------------------------------------------
__global__ void cvt_f32_to_f16(const float* __restrict__ in, f16* __restrict__ out, int n) {
  int i = blockIdx.x * blockDim.x + threadIdx.x;
  if (i < n) out[i] = (f16)in[i];
}

// Transpose + convert: in [R][Ncol] f32  ->  out [Ncol][R] f16  (N-major weights)
__global__ void transpose_cvt(const float* __restrict__ in, f16* __restrict__ out,
                              int R, int Ncol) {
  int idx = blockIdx.x * blockDim.x + threadIdx.x;
  if (idx < R * Ncol) {
    int c = idx / R;      // output row  (original column)
    int r = idx - c * R;  // output col  (original row)
    out[idx] = (f16)in[(size_t)r * Ncol + c];
  }
}

// ---------------------------------------------------------------------------
// Tiled WMMA GEMM:  Cmat[M][N] = A[M][K] * Bt[N][K]^T + bias[N]
// Block: 256 threads = 8 waves, tile 128(M) x 64(N), K stepped by 32.
// Async global->LDS staging, double buffered (3 async ops / wave / stage).
// ---------------------------------------------------------------------------
#define LDS_STRIDE 40   // 32 K-elems + pad (80B rows keep 16B alignment)

template <typename OutT>
__global__ __launch_bounds__(256)
void gemm_wmma(const f16* __restrict__ A, const f16* __restrict__ Bt,
               const float* __restrict__ bias, OutT* __restrict__ Cmat,
               int M, int N, int K) {
  __shared__ f16 sA[2][128 * LDS_STRIDE];
  __shared__ f16 sB[2][64 * LDS_STRIDE];

  const int t    = threadIdx.x;
  const int lane = t & 31;
  const int wave = t >> 5;          // 0..7
  const int wm   = wave & 3;        // 0..3  (M direction)
  const int wn   = wave >> 2;       // 0..1  (N direction)
  const int bm   = blockIdx.x * 128;
  const int bn   = blockIdx.y * 64;

  v8f zero = {};
  v8f acc[2][2];
  acc[0][0] = zero; acc[0][1] = zero; acc[1][0] = zero; acc[1][1] = zero;

  const int lr = t >> 2;            // 0..63 staging row
  const int lc = (t & 3) * 8;       // 0,8,16,24 staging col (8 f16 = 16B)

  const int nsteps = K >> 5;        // K/32

  auto issue = [&](int k0, int buf) {
    async_ld16(lds_addr(&sA[buf][lr * LDS_STRIDE + lc]), A,
               (int)(((bm + lr) * K + k0 + lc) * 2));
    async_ld16(lds_addr(&sA[buf][(lr + 64) * LDS_STRIDE + lc]), A,
               (int)(((bm + lr + 64) * K + k0 + lc) * 2));
    async_ld16(lds_addr(&sB[buf][lr * LDS_STRIDE + lc]), Bt,
               (int)(((bn + lr) * K + k0 + lc) * 2));
  };

  issue(0, 0);
  for (int i = 0; i < nsteps; i++) {
    const int cur = i & 1;
    if (i + 1 < nsteps) { issue((i + 1) << 5, cur ^ 1); WAIT_ASYNC_3(); }
    else                { WAIT_ASYNC_0(); }
    __syncthreads();                       // stage i visible to all waves

    v16h a0 = load_a_frag(&sA[cur][(wm * 32 + 0)  * LDS_STRIDE], LDS_STRIDE, lane);
    v16h a1 = load_a_frag(&sA[cur][(wm * 32 + 16) * LDS_STRIDE], LDS_STRIDE, lane);
    v16h b0 = load_b_frag(&sB[cur][(wn * 32 + 0)  * LDS_STRIDE], LDS_STRIDE, lane);
    v16h b1 = load_b_frag(&sB[cur][(wn * 32 + 16) * LDS_STRIDE], LDS_STRIDE, lane);

    acc[0][0] = wmma_f16(a0, b0, acc[0][0]);
    acc[0][1] = wmma_f16(a0, b1, acc[0][1]);
    acc[1][0] = wmma_f16(a1, b0, acc[1][0]);
    acc[1][1] = wmma_f16(a1, b1, acc[1][1]);
    __syncthreads();                       // all done reading buf[cur]
  }

  // store: D layout -> VGPR r holds M = r + 8*(lane>=16), N = lane&15
  const int col16  = lane & 15;
  const int rowadd = (lane >> 4) * 8;
  for (int i = 0; i < 2; i++) {
    for (int j = 0; j < 2; j++) {
      int c  = bn + wn * 32 + j * 16 + col16;
      int r0 = bm + wm * 32 + i * 16 + rowadd;
      float bv = bias[c];
      #pragma unroll
      for (int r = 0; r < 8; r++) {
        float v = acc[i][j][r] + bv;
        Cmat[(size_t)(r0 + r) * N + c] = (OutT)v;
      }
    }
  }
}

// ---------------------------------------------------------------------------
// Flash attention chunk (32 keys x 16 queries per wave; 2 waves / block share
// the 32-query tile).  MASK only on the diagonal chunk.
// Q fragments are pre-scaled by 0.125*log2(e), so probabilities use exp2.
// ---------------------------------------------------------------------------
template <bool MASK>
__device__ __forceinline__ void attn_chunk(
    int kb, int qrow0, int lane, int wave, int col, int rowadd,
    const f16* __restrict__ kptr, const f16* __restrict__ vptr,
    const v16h (&qf)[2], v8f (&o_acc)[4],
    float (&m)[8], float (&l)[8],
    f16* p_lds, f16* vt) {
  v8f zero = {};

  // ---- S' = (scaled Q) K^T : 4 WMMAs per wave ----
  v16h kf00 = load_b_frag(kptr + (size_t)(kb +  0) * C3 + 0,  C3, lane);
  v16h kf01 = load_b_frag(kptr + (size_t)(kb +  0) * C3 + 32, C3, lane);
  v16h kf10 = load_b_frag(kptr + (size_t)(kb + 16) * C3 + 0,  C3, lane);
  v16h kf11 = load_b_frag(kptr + (size_t)(kb + 16) * C3 + 32, C3, lane);
  v8f s[2];
  s[0] = zero; s[1] = zero;
  s[0] = wmma_f16(qf[0], kf00, s[0]);
  s[0] = wmma_f16(qf[1], kf01, s[0]);
  s[1] = wmma_f16(qf[0], kf10, s[1]);
  s[1] = wmma_f16(qf[1], kf11, s[1]);

  float cmax[8];
  #pragma unroll
  for (int r = 0; r < 8; r++) {
    float s0 = s[0][r];
    float s1 = s[1][r];
    if constexpr (MASK) {
      int q = qrow0 + rowadd + r;
      if (kb + col      > q) s0 = -INFINITY;
      if (kb + 16 + col > q) s1 = -INFINITY;
    }
    s[0][r] = s0; s[1][r] = s1;
    cmax[r] = fmaxf(s0, s1);
  }
  // row max across the 16 lanes holding each row (DPP16 butterfly, VALU only)
  #pragma unroll
  for (int r = 0; r < 8; r++) cmax[r] = row16_max(cmax[r]);

  float alpha[8];
  #pragma unroll
  for (int r = 0; r < 8; r++) {
    float mn = fmaxf(m[r], cmax[r]);
    alpha[r] = exp2f(m[r] - mn);       // exp2(-inf - finite) = 0 on first chunk
    m[r] = mn;
  }

  // P = exp2(S' - m) -> LDS (D-layout -> A-layout crossing), row sums
  const int prow0 = 16 * wave + rowadd;
  float rsum[8];
  #pragma unroll
  for (int r = 0; r < 8; r++) {
    float p0 = exp2f(s[0][r] - m[r]);
    float p1 = exp2f(s[1][r] - m[r]);
    rsum[r] = p0 + p1;
    p_lds[(prow0 + r) * LDS_STRIDE + col]      = (f16)p0;
    p_lds[(prow0 + r) * LDS_STRIDE + 16 + col] = (f16)p1;
  }
  #pragma unroll
  for (int r = 0; r < 8; r++) rsum[r] = row16_sum(rsum[r]);
  #pragma unroll
  for (int r = 0; r < 8; r++) l[r] = l[r] * alpha[r] + rsum[r];

  #pragma unroll
  for (int f = 0; f < 4; f++)
    #pragma unroll
    for (int r = 0; r < 8; r++) o_acc[f][r] *= alpha[r];

  // ---- stage V^T: wave w transposes tokens kb+16w .. kb+16w+15 ----
  // 2 tokens per iteration -> packed 2xf16 (b32) LDS stores.
  const int tkb = 16 * wave;
  #pragma unroll 4
  for (int tk = 0; tk < 16; tk += 2) {
    const f16* vp0 = vptr + (size_t)(kb + tkb + tk) * C3 + 2 * lane;
    const f16* vp1 = vp0 + C3;
    f16 a0 = vp0[0], a1 = vp0[1];
    f16 b0 = vp1[0], b1 = vp1[1];
    v2h pa; pa[0] = a0; pa[1] = b0;        // d=2*lane,   t=tk..tk+1
    v2h pb; pb[0] = a1; pb[1] = b1;        // d=2*lane+1, t=tk..tk+1
    *(v2h*)&vt[(2 * lane)     * LDS_STRIDE + tkb + tk] = pa;
    *(v2h*)&vt[(2 * lane + 1) * LDS_STRIDE + tkb + tk] = pb;
  }
  __syncthreads();                         // P + V^T complete across both waves

  // ---- O += P V : 4 WMMAs per wave ----
  v16h pf = load_a_frag(&p_lds[(16 * wave) * LDS_STRIDE], LDS_STRIDE, lane);
  #pragma unroll
  for (int f = 0; f < 4; f++) {
    v16h vf = load_b_frag(&vt[(f * 16) * LDS_STRIDE], LDS_STRIDE, lane);
    o_acc[f] = wmma_f16(pf, vf, o_acc[f]);
  }
  __syncthreads();                         // done reading before next overwrite
}

// ---------------------------------------------------------------------------
// Flash attention (causal).  Block = 64 threads (2 waves) per 32-query tile;
// wave w owns query rows [qbase+16w, qbase+16w+16).  Diagonal chunk peeled.
// ---------------------------------------------------------------------------
__global__ __launch_bounds__(64)
void flash_attn(const f16* __restrict__ qkv, f16* __restrict__ o) {
  const int tid   = threadIdx.x;
  const int lane  = tid & 31;
  const int wave  = tid >> 5;            // 0..1
  const int qt    = blockIdx.x;          // 0..63 (32-query tiles)
  const int h     = blockIdx.y;          // 0..15
  const int b     = blockIdx.z;          // 0..3
  const int qbase = qt * 32;
  const int qrow0 = qbase + 16 * wave;   // this wave's first query row

  const f16* qptr = qkv + (size_t)b * TT * C3 + (size_t)h * HD;
  const f16* kptr = qptr + CC;
  const f16* vptr = qptr + 2 * CC;

  __shared__ f16 p_lds[32 * LDS_STRIDE];      // P tile 32 x 32, row-major
  __shared__ f16 vt[HD * LDS_STRIDE];         // V^T: [d][token], 32 tokens used

  // Q fragments pre-scaled by 0.125*log2(e) (folds softmax scale + exp->exp2)
  v16h qf[2];
  #pragma unroll
  for (int kd = 0; kd < 2; kd++) {
    v16h tq = load_a_frag(qptr + (size_t)qrow0 * C3 + 32 * kd, C3, lane);
    #pragma unroll
    for (int e = 0; e < 16; e++) tq[e] = tq[e] * (_Float16)QSCALE_LOG2E;
    qf[kd] = tq;
  }

  v8f zero = {};
  v8f o_acc[4];
  #pragma unroll
  for (int f = 0; f < 4; f++) o_acc[f] = zero;

  float m[8], l[8];
  #pragma unroll
  for (int r = 0; r < 8; r++) { m[r] = -INFINITY; l[r] = 0.0f; }

  const int col    = lane & 15;
  const int rowadd = (lane >> 4) * 8;

  const int nch = qt + 1;           // chunks of 32 keys, up to qbase+31
  for (int ch = 0; ch < nch - 1; ch++)
    attn_chunk<false>(ch * 32, qrow0, lane, wave, col, rowadd, kptr, vptr,
                      qf, o_acc, m, l, p_lds, vt);
  attn_chunk<true>((nch - 1) * 32, qrow0, lane, wave, col, rowadd, kptr, vptr,
                   qf, o_acc, m, l, p_lds, vt);

  // normalize and store O (f16) at [b, qrow0+row, h*HD + d]
  f16* optr = o + (size_t)(b * TT + qrow0) * CC + (size_t)h * HD;
  float inv[8];
  #pragma unroll
  for (int r = 0; r < 8; r++) inv[r] = 1.0f / l[r];
  #pragma unroll
  for (int f = 0; f < 4; f++) {
    int d0 = f * 16 + col;
    #pragma unroll
    for (int r = 0; r < 8; r++) {
      float v = o_acc[f][r] * inv[r];
      optr[(size_t)(rowadd + r) * CC + d0] = (f16)v;
    }
  }
}

// ---------------------------------------------------------------------------
// Launch
// ---------------------------------------------------------------------------
extern "C" void kernel_launch(void* const* d_in, const int* in_sizes, int n_in,
                              void* d_out, int out_size, void* d_ws, size_t ws_size,
                              hipStream_t stream) {
  const float* x      = (const float*)d_in[0];   // [B,T,C]
  const float* w_qkv  = (const float*)d_in[1];   // [C, 3C]
  const float* b_qkv  = (const float*)d_in[2];   // [3C]
  const float* w_proj = (const float*)d_in[3];   // [C, C]
  const float* b_proj = (const float*)d_in[4];   // [C]
  float* out = (float*)d_out;                    // [B,T,C]

  // workspace layout (f16 elements)
  f16* ws = (f16*)d_ws;
  f16* xh     = ws;                                    // 8,388,608
  f16* wqkvT  = xh    + (size_t)MROWS * CC;            // 3,145,728  [3C][C]
  f16* wprojT = wqkvT + (size_t)C3 * CC;               // 1,048,576  [C][C]
  f16* qkvh   = wprojT + (size_t)CC * CC;              // 25,165,824 [B*T][3C]
  f16* oh     = qkvh  + (size_t)MROWS * C3;            // 8,388,608  [B*T][C]

  // 1) convert x to f16
  {
    int n = MROWS * CC;
    cvt_f32_to_f16<<<(n + 255) / 256, 256, 0, stream>>>(x, xh, n);
  }
  // 2) transpose+convert weights to N-major f16
  {
    int n = CC * C3;
    transpose_cvt<<<(n + 255) / 256, 256, 0, stream>>>(w_qkv, wqkvT, CC, C3);
  }
  {
    int n = CC * CC;
    transpose_cvt<<<(n + 255) / 256, 256, 0, stream>>>(w_proj, wprojT, CC, CC);
  }
  // 3) QKV GEMM: [8192,1024] x [1024,3072] -> f16 qkv
  {
    dim3 grid(MROWS / 128, C3 / 64);
    gemm_wmma<f16><<<grid, 256, 0, stream>>>(xh, wqkvT, b_qkv, qkvh,
                                             MROWS, C3, CC);
  }
  // 4) causal flash attention (2 waves per 32-query tile, diagonal peeled)
  {
    dim3 grid(TT / 32, HH, BB);
    flash_attn<<<grid, 64, 0, stream>>>(qkvh, oh);
  }
  // 5) projection GEMM: [8192,1024] x [1024,1024] -> f32 out (+bias)
  {
    dim3 grid(MROWS / 128, CC / 64);
    gemm_wmma<float><<<grid, 256, 0, stream>>>(oh, wprojT, b_proj, out,
                                               MROWS, CC, CC);
  }
}
